// RBFDecoder_89215060672773
// MI455X (gfx1250) — matrix operand order repaired and checked by reading, not compiled, not run on hardware
//
#include <hip/hip_runtime.h>

typedef _Float16 v16h __attribute__((ext_vector_type(16)));
typedef float    v8f  __attribute__((ext_vector_type(8)));

#define B_ROWS 100000
#define LBLK   8
#define NRBF   256
#define FDIM   16
#define KTOT   128          // L*F
#define WCOLS  2048         // L*N_RBF
#define NEG_HALF_LOG2E (-0.7213475204444817f)   // -0.5 * log2(e)

// ---------------------------------------------------------------------------
// Prep kernel:
//   bprep[j][col][0..15]  = (f16) exp(beta)
//   bprep[j][col][16..31] = (f16) exp(beta)*center
//   tab[j][col] = float4( -0.5*log2e * term3_prefix, W0, W1, W2 )
// 128 KB + 32 KB -> L2-resident for the whole main kernel.
// ---------------------------------------------------------------------------
__global__ __launch_bounds__(256) void rbf_prep(const float* __restrict__ centers,
                                                const float* __restrict__ betas,
                                                const float* __restrict__ W,
                                                _Float16* __restrict__ bprep,
                                                float4* __restrict__ tab) {
  const int col = threadIdx.x;          // 0..255
  float t3 = 0.f;
  for (int j = 0; j < LBLK; ++j) {
    const int base = (j * NRBF + col) * FDIM;
    _Float16* dst = bprep + (size_t)(j * NRBF + col) * 32;
#pragma unroll
    for (int k = 0; k < FDIM; ++k) {
      float bw = __expf(betas[base + k]);
      float c  = centers[base + k];
      dst[k]      = (_Float16)bw;
      dst[16 + k] = (_Float16)(bw * c);
      t3 += bw * c * c;
    }
    const int wc = j * NRBF + col;
    tab[wc] = make_float4(NEG_HALF_LOG2E * t3,   // prefix over j
                          W[0 * WCOLS + wc], W[1 * WCOLS + wc],
                          W[2 * WCOLS + wc]);
  }
}

// ---------------------------------------------------------------------------
// Main fused kernel: 32 rows per workgroup; 4 waves x (2 M-tiles x 4 N-tiles).
// Each B fragment / table entry is loaded once and reused for both M-tiles.
// Running WMMA accumulators over 8 K-blocks == prefix(term1 - 2*term2).
// ---------------------------------------------------------------------------
__global__ __launch_bounds__(128) void rbf_main(const float* __restrict__ feats,
                                                const _Float16* __restrict__ bprep,
                                                const float4* __restrict__ tab,
                                                const float* __restrict__ bias,
                                                float* __restrict__ out) {
  const int tid  = threadIdx.x;
  const int wave = tid >> 5;
  const int lane = tid & 31;
  const int half = lane >> 4;           // 0: K-lo half, 1: K-hi half
  const int nl   = lane & 15;           // N (and A-row) index within tile
  const int row0 = blockIdx.x * 32;
  const int colbase = wave * 64;

  v8f acc[2][4];
#pragma unroll
  for (int m = 0; m < 2; ++m)
#pragma unroll
    for (int t = 0; t < 4; ++t)
      acc[m][t] = (v8f){0.f,0.f,0.f,0.f,0.f,0.f,0.f,0.f};

  float oacc[2][3][8];
#pragma unroll
  for (int m = 0; m < 2; ++m)
#pragma unroll
    for (int o = 0; o < 3; ++o)
#pragma unroll
      for (int v = 0; v < 8; ++v) oacc[m][o][v] = 0.f;

  const float* frow0 = feats + (size_t)(row0 + nl) * KTOT;
  const float* frow1 = feats + (size_t)(row0 + 16 + nl) * KTOT;

#pragma unroll
  for (int j = 0; j < LBLK; ++j) {
    // ---- A fragments (2 M-tiles): lane(h,m) holds K=8h..8h+7 (x^2), 16+8h.. (-2x)
    v16h a[2];
#pragma unroll
    for (int m = 0; m < 2; ++m) {
      const float4* fp =
          (const float4*)((m ? frow1 : frow0) + j * 16 + half * 8);
      float4 p0 = fp[0];
      float4 p1 = fp[1];
      float xs[8] = {p0.x, p0.y, p0.z, p0.w, p1.x, p1.y, p1.z, p1.w};
#pragma unroll
      for (int e = 0; e < 8; ++e) {
        a[m][e]     = (_Float16)(xs[e] * xs[e]);   // term1 slice
        a[m][8 + e] = (_Float16)(-2.f * xs[e]);    // -2*term2 slice
      }
    }

    // ---- 8 WMMAs: each B fragment loaded once, used by both M-tiles
#pragma unroll
    for (int t = 0; t < 4; ++t) {
      const int col = colbase + t * 16 + nl;
      const _Float16* bp =
          bprep + (size_t)(j * NRBF + col) * 32 + half * 8;
      v16h bfrag;
#pragma unroll
      for (int e = 0; e < 8; ++e) {
        bfrag[e]     = bp[e];
        bfrag[8 + e] = bp[16 + e];
      }
      acc[0][t] = __builtin_amdgcn_wmma_f32_16x16x32_f16(
          false, a[0], false, bfrag, (short)0, acc[0][t], false, false);
      acc[1][t] = __builtin_amdgcn_wmma_f32_16x16x32_f16(
          false, a[1], false, bfrag, (short)0, acc[1][t], false, false);
    }

    // ---- consume prefix-j snapshot: rbf = 2^(acc*c + t3c), fold into out
    //      native v_exp_f32: argument is in [-O(10), 0], far from the
    //      subnormal boundary, so no libm denormal fix-up is needed.
#pragma unroll
    for (int t = 0; t < 4; ++t) {
      const int col = colbase + t * 16 + nl;
      const float4 tw = tab[j * NRBF + col];   // {t3c, w0, w1, w2}
#pragma unroll
      for (int m = 0; m < 2; ++m) {
#pragma unroll
        for (int v = 0; v < 8; ++v) {
          float r = __builtin_amdgcn_exp2f(
              fmaf(acc[m][t][v], NEG_HALF_LOG2E, tw.x));
          oacc[m][0][v] += r * tw.y;
          oacc[m][1][v] += r * tw.z;
          oacc[m][2][v] += r * tw.w;
        }
      }
    }
  }

  // ---- reduce over the 16 N-lanes of each half (rows = mt*16 + 8*half + v)
#pragma unroll
  for (int m = 0; m < 2; ++m)
#pragma unroll
    for (int o = 0; o < 3; ++o)
#pragma unroll
      for (int v = 0; v < 8; ++v) {
        float s = oacc[m][o][v];
        s += __shfl_xor(s, 1, 32);
        s += __shfl_xor(s, 2, 32);
        s += __shfl_xor(s, 4, 32);
        s += __shfl_xor(s, 8, 32);
        oacc[m][o][v] = s;
      }

  __shared__ float red[4][32][3];
  if (nl == 0) {
#pragma unroll
    for (int m = 0; m < 2; ++m)
#pragma unroll
      for (int v = 0; v < 8; ++v)
#pragma unroll
        for (int o = 0; o < 3; ++o)
          red[wave][m * 16 + half * 8 + v][o] = oacc[m][o][v];
  }
  __syncthreads();

  if (tid < 96) {
    const int r = tid / 3, o = tid - r * 3;
    float s = red[0][r][o] + red[1][r][o] + red[2][r][o] + red[3][r][o] +
              bias[o];
    out[(size_t)(row0 + r) * 3 + o] = s;
  }
}

// ---------------------------------------------------------------------------
extern "C" void kernel_launch(void* const* d_in, const int* in_sizes, int n_in,
                              void* d_out, int out_size, void* d_ws, size_t ws_size,
                              hipStream_t stream) {
  // inputs: x(unused), feats, centers, betas, W, b
  const float* feats   = (const float*)d_in[1];
  const float* centers = (const float*)d_in[2];
  const float* betas   = (const float*)d_in[3];
  const float* W       = (const float*)d_in[4];
  const float* b       = (const float*)d_in[5];
  float* out = (float*)d_out;

  _Float16* bprep = (_Float16*)d_ws;                          // 131072 B
  float4* tab = (float4*)((char*)d_ws + (size_t)LBLK * NRBF * 32 * sizeof(_Float16)); // 32768 B

  rbf_prep<<<1, 256, 0, stream>>>(centers, betas, W, bprep, tab);
  rbf_main<<<B_ROWS / 32, 128, 0, stream>>>(feats, bprep, tab, b, out);
}
